// GraphSage_57509612093516
// MI455X (gfx1250) — matrix operand order repaired and checked by reading, not compiled, not run on hardware
//
#include <hip/hip_runtime.h>
#include <stdint.h>

#define NN 50000
#define NE 400000
#define DD 512
#define NL 3
#define NP 50016   // rows padded to multiple of 32 (1563 * 32)

typedef __bf16 v16bf __attribute__((ext_vector_type(16)));
typedef float  v8f   __attribute__((ext_vector_type(8)));

union BF16x16 {
    uint4 q[2];
    v16bf v;
};

__device__ __forceinline__ unsigned short bf16_bits(float f) {
    union { __bf16 b; unsigned short u; } c;
    c.b = (__bf16)f;
    return c.u;
}

__device__ __forceinline__ float bf16_val(unsigned short u) {
    union { unsigned short u; __bf16 b; } c;
    c.u = u;
    return (float)c.b;
}

// ---------------- utility kernels ----------------

__global__ void zero_f32(float* __restrict__ p, int n) {
    int i = blockIdx.x * blockDim.x + threadIdx.x;
    if (i < n) p[i] = 0.0f;
}

__global__ void count_deg(const int* __restrict__ dst, float* __restrict__ deg) {
    int e = blockIdx.x * blockDim.x + threadIdx.x;
    if (e < NE) atomicAdd(&deg[dst[e]], 1.0f);
}

__global__ void invert_deg(float* __restrict__ deg) {
    int i = blockIdx.x * blockDim.x + threadIdx.x;
    if (i < NN) {
        float d = deg[i];
        deg[i] = (d > 0.0f) ? (1.0f / d) : 0.0f;
    }
}

// W: [D][D] row-major (k, n). Wt: [n][k] bf16 -> each WMMA B lane reads
// 16 consecutive K values of one output column.
__global__ void pack_w(const float* __restrict__ W, unsigned short* __restrict__ Wt) {
    int i = blockIdx.x * blockDim.x + threadIdx.x;
    if (i < DD * DD) {
        int n = i / DD, k = i % DD;
        Wt[i] = bf16_bits(W[k * DD + n]);
    }
}

// plain f32 -> bf16 row mirror (for x at layer 0)
__global__ void cvt_f32_bf16(const float* __restrict__ in,
                             unsigned short* __restrict__ out, int n) {
    int i = 4 * (blockIdx.x * blockDim.x + threadIdx.x);
    if (i < n) {
        float4 v = *(const float4*)(in + i);
        ushort4 o;
        o.x = bf16_bits(v.x); o.y = bf16_bits(v.y);
        o.z = bf16_bits(v.z); o.w = bf16_bits(v.w);
        *(ushort4*)(out + i) = o;
    }
}

// agg (f32) * inv_deg[row] -> bf16 mirror (mean aggregation folded in)
__global__ void cvt_agg_bf16(const float* __restrict__ agg,
                             const float* __restrict__ inv_deg,
                             unsigned short* __restrict__ out) {
    int i = 4 * (blockIdx.x * blockDim.x + threadIdx.x);
    if (i < NN * DD) {
        float s = inv_deg[i / DD];
        float4 v = *(const float4*)(agg + i);
        ushort4 o;
        o.x = bf16_bits(v.x * s); o.y = bf16_bits(v.y * s);
        o.z = bf16_bits(v.z * s); o.w = bf16_bits(v.w * s);
        *(ushort4*)(out + i) = o;
    }
}

// One block per edge; 128 threads x 4 bf16 = 512 features.
// Gather from bf16 activations (halves gather bandwidth), f32 atomic accumulate.
__global__ __launch_bounds__(128) void scatter_edges(
    const unsigned short* __restrict__ zb, const int* __restrict__ src,
    const int* __restrict__ dst, const float* __restrict__ w,
    float* __restrict__ agg)
{
    int e = blockIdx.x;
    int t = threadIdx.x;
    int s = src[e];      // uniform within block -> scalar load
    int d = dst[e];
    float ww = w[e];
    union { uint2 p; unsigned short u[4]; } v;
    v.p = ((const uint2*)(zb + (size_t)s * DD))[t];
    float* ap = agg + (size_t)d * DD + (size_t)t * 4;
    atomicAdd(ap + 0, bf16_val(v.u[0]) * ww);
    atomicAdd(ap + 1, bf16_val(v.u[1]) * ww);
    atomicAdd(ap + 2, bf16_val(v.u[2]) * ww);
    atomicAdd(ap + 3, bf16_val(v.u[3]) * ww);
}

// z_out = relu( agg_scaled @ Wl + z_in @ Wr + bias )
// grid = NP/32 blocks, 256 threads (8 waves). Wave w: rows [32*bid, +32) as
// two 16-row tiles, cols [64*w, +64) as 4 n-tiles -> 8 f32 accumulators,
// 16 v_wmma_f32_16x16x32_bf16 per K-step. All inputs already bf16: the
// inner loop is pure global_load_b128 + wmma (zero conversion VALU).
__global__ __launch_bounds__(256) void sage_gemm(
    const unsigned short* __restrict__ zb,     // [NP][DD] bf16
    const unsigned short* __restrict__ ab,     // [NP][DD] bf16 (scaled agg)
    const unsigned short* __restrict__ Wlt,    // [DD][DD] bf16, n-major
    const unsigned short* __restrict__ Wrt,
    const float* __restrict__ bias,
    float* __restrict__ outf,                  // final layer: f32 output
    unsigned short* __restrict__ outb,         // hidden layers: bf16 output
    int write_f32)
{
    const int lane = threadIdx.x & 31;
    const int wave = threadIdx.x >> 5;
    const int m0 = blockIdx.x * 32;
    const int n0 = wave * 64;

    // A 16x32 as loaded here (contiguous k, same order as B): lanes 0-15 row=lane, K {0..15};
    // lanes 16-31 row=lane-16, K {16..31}.
    const int mrow = lane & 15;
    const int kb   = (lane >> 4) * 16;
    // B layout: lane holds one column (lane&15), 16 consecutive K values.
    const int ncol = lane & 15;
    const int ksub = (lane >> 4) * 16;

    const unsigned short* paA[2];   // scaled agg rows
    const unsigned short* paZ[2];   // z rows
    #pragma unroll
    for (int mt = 0; mt < 2; ++mt) {
        const size_t row = (size_t)(m0 + mt * 16 + mrow);
        paA[mt] = ab + row * DD + kb;
        paZ[mt] = zb + row * DD + kb;
    }

    v8f acc[2][4] = {};

    for (int k0 = 0; k0 < DD; k0 += 32) {
        __builtin_prefetch((const char*)(paA[0] + k0) + 64, 0, 1);
        __builtin_prefetch((const char*)(paZ[0] + k0) + 64, 0, 1);

        BF16x16 Aa[2], Az[2];
        #pragma unroll
        for (int mt = 0; mt < 2; ++mt) {
            const uint4* pa = (const uint4*)(paA[mt] + k0);  // 8 bf16 per uint4
            const uint4* pz = (const uint4*)(paZ[mt] + k0);
            Aa[mt].q[0] = pa[0]; Aa[mt].q[1] = pa[1];        // K +0..15
            Az[mt].q[0] = pz[0]; Az[mt].q[1] = pz[1];
        }

        #pragma unroll
        for (int nt = 0; nt < 4; ++nt) {
            const int n = n0 + nt * 16 + ncol;
            const uint4* bl = (const uint4*)(Wlt + (size_t)n * DD + k0 + ksub);
            const uint4* br = (const uint4*)(Wrt + (size_t)n * DD + k0 + ksub);
            BF16x16 Bl, Br;
            Bl.q[0] = bl[0]; Bl.q[1] = bl[1];
            Br.q[0] = br[0]; Br.q[1] = br[1];
            #pragma unroll
            for (int mt = 0; mt < 2; ++mt) {
                acc[mt][nt] = __builtin_amdgcn_wmma_f32_16x16x32_bf16(
                    false, Aa[mt].v, false, Bl.v, (short)0, acc[mt][nt], false, false);
                acc[mt][nt] = __builtin_amdgcn_wmma_f32_16x16x32_bf16(
                    false, Az[mt].v, false, Br.v, (short)0, acc[mt][nt], false, false);
            }
        }
    }

    // Epilogue: bias + relu + guarded store.
    // C/D layout: VGPR r, lane l -> row = m0 + mt*16 + r + 8*(l>>4), col = n0t + (l&15)
    #pragma unroll
    for (int mt = 0; mt < 2; ++mt) {
        const int rbase = m0 + mt * 16 + ((lane >> 4) << 3);
        #pragma unroll
        for (int nt = 0; nt < 4; ++nt) {
            const int col = n0 + nt * 16 + (lane & 15);
            const float bv = bias[col];
            #pragma unroll
            for (int r = 0; r < 8; ++r) {
                const int row = rbase + r;
                float vv = acc[mt][nt][r] + bv;
                vv = vv > 0.0f ? vv : 0.0f;
                if (row < NN) {
                    if (write_f32) outf[(size_t)row * DD + col] = vv;
                    else           outb[(size_t)row * DD + col] = bf16_bits(vv);
                }
            }
        }
    }
}

// ---------------- host launch ----------------

extern "C" void kernel_launch(void* const* d_in, const int* in_sizes, int n_in,
                              void* d_out, int out_size, void* d_ws, size_t ws_size,
                              hipStream_t stream) {
    (void)in_sizes; (void)n_in; (void)out_size; (void)ws_size;
    const float* x  = (const float*)d_in[0];
    const int*   ei = (const int*)d_in[1];        // [2, E]: src then dst
    const float* ew = (const float*)d_in[2];
    const float* Wl = (const float*)d_in[3];      // [L, D, D]
    const float* Wr = (const float*)d_in[4];      // [L, D, D]
    const float* b  = (const float*)d_in[5];      // [L, D]
    float* out = (float*)d_out;

    const int* src = ei;
    const int* dst = ei + NE;

    // workspace layout
    char* ws = (char*)d_ws;
    const size_t aggB = (size_t)NN * DD * sizeof(float);            // 102.4 MB
    const size_t zbB  = (size_t)NP * DD * sizeof(unsigned short);   // 51.2 MB each
    float*          agg    = (float*)(ws);
    unsigned short* ab     = (unsigned short*)(ws + aggB);
    unsigned short* zb0    = (unsigned short*)(ws + aggB + zbB);
    unsigned short* zb1    = (unsigned short*)(ws + aggB + 2 * zbB);
    float*          invdeg = (float*)(ws + aggB + 3 * zbB);
    size_t degB = ((size_t)NN * sizeof(float) + 255) & ~(size_t)255;
    unsigned short* Wt     = (unsigned short*)(ws + aggB + 3 * zbB + degB);
    // Wt layout: [Wl layer0..2 | Wr layer0..2], each D*D bf16

    // degree -> inverse degree
    zero_f32<<<(NN + 255) / 256, 256, 0, stream>>>(invdeg, NN);
    count_deg<<<(NE + 255) / 256, 256, 0, stream>>>(dst, invdeg);
    invert_deg<<<(NN + 255) / 256, 256, 0, stream>>>(invdeg);

    // pack weights to bf16 transposed (B-operand friendly)
    const int wgrid = (DD * DD + 255) / 256;
    for (int l = 0; l < NL; ++l) {
        pack_w<<<wgrid, 256, 0, stream>>>(Wl + (size_t)l * DD * DD, Wt + (size_t)l * DD * DD);
        pack_w<<<wgrid, 256, 0, stream>>>(Wr + (size_t)l * DD * DD, Wt + (size_t)(NL + l) * DD * DD);
    }

    // x -> bf16 mirror
    const int elemN = NN * DD;
    cvt_f32_bf16<<<(elemN / 4 + 255) / 256, 256, 0, stream>>>(x, zb0, elemN);

    for (int l = 0; l < NL; ++l) {
        const unsigned short* zin = (l == 1) ? zb1 : zb0;  // 0:zb0 1:zb1 2:zb0
        unsigned short* zoutb     = (l == 0) ? zb1 : zb0;
        const int last = (l == NL - 1);

        zero_f32<<<(elemN + 255) / 256, 256, 0, stream>>>(agg, elemN);
        scatter_edges<<<NE, 128, 0, stream>>>(zin, src, dst, ew, agg);
        cvt_agg_bf16<<<(elemN / 4 + 255) / 256, 256, 0, stream>>>(agg, invdeg, ab);
        sage_gemm<<<NP / 32, 256, 0, stream>>>(
            zin, ab,
            Wt + (size_t)l * DD * DD, Wt + (size_t)(NL + l) * DD * DD,
            b + (size_t)l * DD,
            last ? out : nullptr, last ? nullptr : zoutb, last);
    }
}
